// Transformer_27298812133999
// MI455X (gfx1250) — matrix operand order repaired and checked by reading, not compiled, not run on hardware
//
// CDNA5 (gfx1250 / MI455X) enc-dec transformer forward.
//
// Key optimizations (reasoned from MI455X specs, compile-only):
//  * bf16 WMMA (v_wmma_f32_16x16x32_bf16) with f32 accumulation — workload is
//    compute-bound (~370 GFLOP vs ~10us of HBM traffic at 23.3 TB/s).
//  * All large linear GEMMs stage A/B tiles into LDS via the Tensor Data
//    Mover (tensor_load_to_lds, TENSORcnt), double-buffered so the DMA of
//    tile k+1 overlaps the WMMAs of tile k. Weights are converted ONCE per
//    launch to bf16, pre-transposed to [N][K] so TDM tiles land exactly in
//    the fragment-friendly k-contiguous LDS layout (ds_load_b128 -> wmma).
//    The LDS array's address is escaped via inline asm so the compiler keeps
//    the fragment loads (TDM writes LDS behind the compiler's back).
//  * Encoder self-attn and cross-attn have NO softmax in the reference, so
//    (QK^T/sqrt(hd))V reassociates to Q(K^T V)/sqrt(hd): per-head 64x64
//    inner product — removes all O(T^2) work and any T x T tensor.
//  * Decoder causal self-attn keeps a full 64x1024 score strip in 264KB of
//    the WGP's 320KB LDS: exact softmax, zero global scratch, WMMA for both
//    QK^T and P*V; above-diagonal key tiles skipped.
//
// Input flattening assumption: setup_inputs() dict insertion order, nested
// dicts depth-first in insertion order:
//  0 encoder_seq, 1 decoder_seq,
//  2 tok_emb_enc, 3 pos_emb_enc, 4 tok_emb_dec, 5 pos_emb_dec,
//  6..13  enc_attn {Wq,bq,Wk,bk,Wv,bv,ln_g,ln_b},
//  14..19 enc_ffn  {W1,b1,W2,b2,ln_g,ln_b},
//  20..27 dec_self, 28..35 dec_cross, 36..41 dec_ffn,
//  42 W_out, 43 b_out.
//
// Workspace: ~50MB fp32 + ~190MB bf16 (weights + activation mirrors) ~ 240MB.

#include <hip/hip_runtime.h>
#include <hip/hip_bf16.h>
#include <cstdint>
#include <cstddef>

// ---------------------------------------------------------------- constants
constexpr int Dm   = 1024;   // model dim
constexpr int Hh   = 16;     // heads
constexpr int HD   = 64;     // head dim
constexpr int TE   = 1024;   // encoder tokens
constexpr int TD   = 1024;   // decoder tokens
constexpr int Bb   = 2;      // batch
constexpr int Lc   = 2;      // layers per stack
constexpr int DFF  = 4096;   // ffn hidden
constexpr int VD   = 32000;  // decoder vocab
constexpr int ROWS_E = Bb * TE;   // 2048
constexpr int ROWS_D = Bb * TD;   // 2048

constexpr int SROW = 1032;                    // padded LDS score row (floats)
constexpr size_t ATTN_SMEM = (size_t)64 * SROW * sizeof(float);  // 264,192 B

// ---------------------------------------------------------------- wmma types
typedef __attribute__((ext_vector_type(16))) __bf16 v16bf;
typedef __attribute__((ext_vector_type(8)))  float  v8f;
typedef __attribute__((ext_vector_type(4)))  unsigned int u32x4;
typedef __attribute__((ext_vector_type(8)))  int i32x8;
typedef __attribute__((ext_vector_type(4)))  int i32x4;

union FragBF { v16bf v; unsigned u[8]; };

__device__ inline unsigned short f2bf(float x) {
  unsigned u = __builtin_bit_cast(unsigned, x);
  unsigned r = (u + 0x7FFFu + ((u >> 16) & 1u)) >> 16;   // RNE
  return (unsigned short)r;
}
__device__ inline unsigned packbf(float lo, float hi) {
  return (unsigned)f2bf(lo) | ((unsigned)f2bf(hi) << 16);
}
__device__ inline v8f vzero8() {
  v8f z = {0.f, 0.f, 0.f, 0.f, 0.f, 0.f, 0.f, 0.f};
  return z;
}
__device__ inline v8f wmma_bf16(FragBF a, FragBF b, v8f c) {
  return __builtin_amdgcn_wmma_f32_16x16x32_bf16(
      /*neg_a=*/false, a.v, /*neg_b=*/false, b.v,
      /*c_mod=*/(short)0, c, /*reuse_a=*/false, /*reuse_b=*/false);
}

// ----------------------------------------------------------- TDM tile load
// Build a 2D Tensor-DMA descriptor (ISA cdna5 ch.8) for a bf16 tile of
// `rows` rows x 64 (BK) cols out of a packed [rows_total][Kst] bf16 tensor,
// landing row-major/k-contiguous in LDS at byte offset ldsOff.
// data_size=1 (2B), type=2 ("image"), count=1, no iterate/pad/multicast.
// This toolchain's builtin is the 6-arg form:
//   (u32x4 g0, i32x8 g1, i32x4 g2, i32x4 g3, i32x8 g4, i32 cpol)
__device__ inline void tdm_tile(const unsigned short* gsrc, unsigned ldsOff,
                                int rows, int Kst) {
  unsigned long long ga = (unsigned long long)(uintptr_t)gsrc;
  u32x4 g0;
  g0[0] = 1u;                                           // count=1, load
  g0[1] = ldsOff;                                       // lds_addr
  g0[2] = (unsigned)ga;                                 // global_addr[31:0]
  g0[3] = (unsigned)((ga >> 32) & 0x01FFFFFFu) | (2u << 30);  // [56:32]|type=2
  i32x8 g1;
  g1[0] = (int)(1u << 16);                              // data_size=1 (2B)
  g1[1] = (int)(((unsigned)Kst & 0xFFFFu) << 16);       // tensor_dim0[15:0]
  g1[2] = (int)(((((unsigned)Kst) >> 16) & 0xFFFFu) |
                (((unsigned)rows & 0xFFFFu) << 16));    // dim0[31:16]|dim1[15:0]
  g1[3] = (int)(((((unsigned)rows) >> 16) & 0xFFFFu) |
                (64u << 16));                           // dim1[31:16]|tile_dim0=64
  g1[4] = (int)rows;                                    // tile_dim1; tile_dim2=0
  g1[5] = (int)Kst;                                     // dim0_stride[31:0]
  g1[6] = 0;                                            // stride hi / dim1_stride
  g1[7] = 0;
  i32x4 z4 = {0, 0, 0, 0};                              // 2D: groups 2/3 unused
  i32x8 z8 = {0, 0, 0, 0, 0, 0, 0, 0};
  __builtin_amdgcn_tensor_load_to_lds(g0, g1, z4, z4, z8, 0);
}

// ------------------------------------------------------------------ embed
__global__ __launch_bounds__(256) void embed_kernel(
    const int* __restrict__ seq, const float* __restrict__ tok,
    const float* __restrict__ pos, float* __restrict__ x,
    unsigned short* __restrict__ xb, int T, int D) {
  int i = blockIdx.x;            // flat row 0..B*T-1
  int t = i % T;
  int id = seq[i];
  for (int d = threadIdx.x; d < D; d += blockDim.x) {
    float vv = tok[(size_t)id * D + d] + pos[(size_t)t * D + d];
    x[(size_t)i * D + d] = vv;
    xb[(size_t)i * D + d] = f2bf(vv);
  }
}

// -------------------------------------------- weight transpose+convert (once)
// w: [K][N] fp32  ->  wt: [N][K] bf16   (LDS-tiled for coalescing both sides)
__global__ __launch_bounds__(256) void transpose_convert_kernel(
    const float* __restrict__ w, unsigned short* __restrict__ wt,
    int K, int N) {
  __shared__ float tile[32][33];
  int kb = blockIdx.y * 32, nb = blockIdx.x * 32;
  int tx = threadIdx.x & 31, ty = threadIdx.x >> 5;   // 8 row-groups
  for (int r = ty; r < 32; r += 8)
    tile[r][tx] = w[(size_t)(kb + r) * N + nb + tx];
  __syncthreads();
  for (int r = ty; r < 32; r += 8)
    wt[(size_t)(nb + r) * K + kb + tx] = f2bf(tile[tx][r]);
}

#define BM 128
#define BN 128
#define BK 64

// ----------------------------- TDM-staged bf16 GEMM (all big linear layers)
// C = A[M][K](bf16) x Bt[N][K]^T(bf16) (+bias)(+relu); fp32 and/or bf16 out.
// Double-buffered LDS; wave 0 drives the Tensor Data Mover, prefetching the
// next K-tile while all 8 waves run WMMAs on the current one.
// NOTE: assumes this kernel's static LDS block starts at LDS offset 0
// (single __shared__ object), since TDM descriptors take raw LDS addresses.
// M,N multiples of 128; K multiple of 64 (true for every call site).
__global__ __launch_bounds__(256) void gemm_bf16_tdm(
    const unsigned short* __restrict__ A,   // [M][K] bf16
    const unsigned short* __restrict__ Bt,  // [N][K] bf16 (pre-transposed)
    const float* __restrict__ bias,
    float* __restrict__ C,                  // nullable
    unsigned short* __restrict__ Cb,        // nullable bf16 mirror
    int M, int N, int K, int relu) {
  __shared__ unsigned short smem[2 * (BM * BK + BN * BK)];   // 64 KB
  // ushort-index layout: A0 @0, A1 @BM*BK, B0 @2*BM*BK, B1 @2*BM*BK+BN*BK
  const unsigned aByte[2] = {0u, (unsigned)(BM * BK * 2)};
  const unsigned bByte[2] = {(unsigned)(2 * BM * BK * 2),
                             (unsigned)(2 * BM * BK * 2 + BN * BK * 2)};

  // Escape smem's address: the TDM engine writes it behind the compiler's
  // back, so without this LLVM treats the array as store-free and folds the
  // fragment loads to undef. Input-only operand keeps LDS provenance so the
  // loads still lower to ds_load_b128.
  asm volatile("" ::"v"((unsigned short*)smem) : "memory");

  int n0 = blockIdx.x * BN, m0 = blockIdx.y * BM;
  int tid = threadIdx.x, wid = tid >> 5, lane = tid & 31;
  int wm = wid & 1, wn = wid >> 1;        // 2x4 wave grid; wave tile 64x32
  int lm = lane & 15, half = lane >> 4;

  v8f acc[4][2];
#pragma unroll
  for (int mt = 0; mt < 4; ++mt)
#pragma unroll
    for (int nt = 0; nt < 2; ++nt) acc[mt][nt] = vzero8();

  if (wid == 0) {                          // prologue: DMA first tiles
    tdm_tile(A + (size_t)m0 * K, aByte[0], BM, K);
    tdm_tile(Bt + (size_t)n0 * K, bByte[0], BN, K);
  }

  int cur = 0;
  for (int k0 = 0; k0 < K; k0 += BK) {
    if (wid == 0) __builtin_amdgcn_s_wait_tensorcnt(0);
    __syncthreads();                       // tiles[cur] visible to all waves
    asm volatile("" ::: "memory");         // force re-read of DMA'd LDS
    if (wid == 0 && k0 + BK < K) {         // prefetch next K-tile
      tdm_tile(A + (size_t)m0 * K + (k0 + BK), aByte[cur ^ 1], BM, K);
      tdm_tile(Bt + (size_t)n0 * K + (k0 + BK), bByte[cur ^ 1], BN, K);
    }
    const unsigned short* As = &smem[cur ? BM * BK : 0];
    const unsigned short* Bs = &smem[2 * BM * BK + (cur ? BN * BK : 0)];
#pragma unroll
    for (int ks = 0; ks < 2; ++ks) {       // two K=32 WMMA steps per tile
      FragBF bfr[2];
#pragma unroll
      for (int nt = 0; nt < 2; ++nt) {
        const unsigned* bp =
            (const unsigned*)&Bs[(wn * 32 + nt * 16 + lm) * BK + ks * 32];
#pragma unroll
        for (int r = 0; r < 8; ++r) bfr[nt].u[r] = bp[half * 8 + r];
      }
#pragma unroll
      for (int mt = 0; mt < 4; ++mt) {
        FragBF afr;
        const unsigned* ap =
            (const unsigned*)&As[(wm * 64 + mt * 16 + lm) * BK + ks * 32];
#pragma unroll
        for (int r = 0; r < 8; ++r)
          afr.u[r] = ap[half * 8 + r];
#pragma unroll
        for (int nt = 0; nt < 2; ++nt)
          acc[mt][nt] = wmma_bf16(afr, bfr[nt], acc[mt][nt]);
      }
    }
    cur ^= 1;
  }

  // epilogue: bias, relu, fp32 and/or bf16 stores (exact tiling, no guards)
#pragma unroll
  for (int mt = 0; mt < 4; ++mt) {
#pragma unroll
    for (int nt = 0; nt < 2; ++nt) {
      int gn = n0 + wn * 32 + nt * 16 + lm;
#pragma unroll
      for (int r = 0; r < 8; ++r) {
        int gm = m0 + wm * 64 + mt * 16 + r + half * 8;
        float vv = acc[mt][nt][r];
        if (bias) vv += bias[gn];
        if (relu) vv = fmaxf(vv, 0.f);
        if (C)  C[(size_t)gm * N + gn] = vv;
        if (Cb) Cb[(size_t)gm * N + gn] = f2bf(vv);
      }
    }
  }
}

// ----------------------- generic fp32-staged WMMA GEMM (tiny head-batched ops)
// Used only for the reassociated linear attention: K^T V and Q(K^T V).
__global__ __launch_bounds__(256) void gemm_wmma(
    const float* __restrict__ A, const float* __restrict__ Bw,
    const float* __restrict__ bias, float* __restrict__ C,
    int M, int N, int K, int lda, int ldb, int ldc,
    long long bsA, long long bsAh, long long bsB, long long bsBh,
    long long bsC, long long bsCh, int Hdiv,
    int transA, int transB, float alpha, int relu) {
  __shared__ unsigned short As[BM * BK];   // bf16 bits, row-major [m][k]
  __shared__ unsigned short Bs[BN * BK];   // bf16 bits, k-contiguous [n][k]

  int bz = blockIdx.z;
  int bo = bz / Hdiv, bh = bz % Hdiv;
  A  += bo * bsA + bh * bsAh;
  Bw += bo * bsB + bh * bsBh;
  C  += bo * bsC + bh * bsCh;

  int n0 = blockIdx.x * BN;
  int m0 = blockIdx.y * BM;
  int tid = threadIdx.x;
  int wid = tid >> 5, lane = tid & 31;
  int wm = wid & 1, wn = wid >> 1;
  int lm = lane & 15, half = lane >> 4;

  v8f acc[4][2];
#pragma unroll
  for (int mt = 0; mt < 4; ++mt)
#pragma unroll
    for (int nt = 0; nt < 2; ++nt) acc[mt][nt] = vzero8();

  for (int k0 = 0; k0 < K; k0 += BK) {
    for (int idx = tid; idx < BM * BK; idx += 256) {
      int i = idx >> 6, j = idx & (BK - 1);
      int gm = m0 + i, gk = k0 + j;
      float va = 0.f;
      if (gm < M && gk < K)
        va = transA ? A[(size_t)gk * lda + gm] : A[(size_t)gm * lda + gk];
      As[idx] = f2bf(va);
    }
    for (int idx = tid; idx < BK * BN; idx += 256) {
      int kk = idx >> 7, n = idx & (BN - 1);
      int gk = k0 + kk, gn = n0 + n;
      float vb = 0.f;
      if (gk < K && gn < N)
        vb = transB ? Bw[(size_t)gn * ldb + gk] : Bw[(size_t)gk * ldb + gn];
      Bs[n * BK + kk] = f2bf(vb);
    }
    __syncthreads();
#pragma unroll
    for (int ks = 0; ks < 2; ++ks) {
      FragBF bfr[2];
#pragma unroll
      for (int nt = 0; nt < 2; ++nt) {
        const unsigned* bp =
            (const unsigned*)&Bs[(wn * 32 + nt * 16 + lm) * BK + ks * 32];
#pragma unroll
        for (int r = 0; r < 8; ++r) bfr[nt].u[r] = bp[half * 8 + r];
      }
#pragma unroll
      for (int mt = 0; mt < 4; ++mt) {
        FragBF afr;
        const unsigned* ap =
            (const unsigned*)&As[(wm * 64 + mt * 16 + lm) * BK + ks * 32];
#pragma unroll
        for (int r = 0; r < 8; ++r)
          afr.u[r] = ap[half * 8 + r];
#pragma unroll
        for (int nt = 0; nt < 2; ++nt)
          acc[mt][nt] = wmma_bf16(afr, bfr[nt], acc[mt][nt]);
      }
    }
    __syncthreads();
  }
#pragma unroll
  for (int mt = 0; mt < 4; ++mt) {
#pragma unroll
    for (int nt = 0; nt < 2; ++nt) {
      int gn = n0 + wn * 32 + nt * 16 + lm;
#pragma unroll
      for (int r = 0; r < 8; ++r) {
        int gm = m0 + wm * 64 + mt * 16 + r + half * 8;
        if (gm < M && gn < N) {
          float vv = alpha * acc[mt][nt][r];
          if (bias) vv += bias[gn];
          if (relu) vv = fmaxf(vv, 0.f);
          C[(size_t)gm * ldc + gn] = vv;
        }
      }
    }
  }
}

// ------------------------------------- decoder causal self-attn (scores in LDS)
__global__ __launch_bounds__(128) void attn_softmax_causal(
    const float* __restrict__ q, const float* __restrict__ k,
    const float* __restrict__ v, float* __restrict__ out, int T, int D) {
  extern __shared__ float S[];           // [64][SROW]
  int b = blockIdx.z, h = blockIdx.y;
  int q0 = blockIdx.x * 64;
  const float* qp = q + (size_t)b * T * D + h * HD;
  const float* kp = k + (size_t)b * T * D + h * HD;
  const float* vp = v + (size_t)b * T * D + h * HD;
  float*       op = out + (size_t)b * T * D + h * HD;
  int tid = threadIdx.x, wid = tid >> 5, lane = tid & 31;
  int lm = lane & 15, half = lane >> 4;
  int NT = (q0 >> 4) + 4;                // key tiles at/below the diagonal

  FragBF aq[4][2];
#pragma unroll
  for (int mt = 0; mt < 4; ++mt)
#pragma unroll
    for (int ks = 0; ks < 2; ++ks) {
      const float* qr = qp + (size_t)(q0 + mt * 16 + lm) * D + ks * 32;
#pragma unroll
      for (int r = 0; r < 8; ++r) {
        int kk = half * 16 + 2 * r;
        aq[mt][ks].u[r] = packbf(qr[kk], qr[kk + 1]);
      }
    }

  for (int nt = wid; nt < NT; nt += 4) {   // Phase 1: S = (Q K^T)/8
    v8f acc[4];
#pragma unroll
    for (int mt = 0; mt < 4; ++mt) acc[mt] = vzero8();
#pragma unroll
    for (int ks = 0; ks < 2; ++ks) {
      FragBF bk_;
      const float* kr = kp + (size_t)(nt * 16 + lm) * D + ks * 32;
#pragma unroll
      for (int r = 0; r < 8; ++r) {
        int kk = half * 16 + 2 * r;
        bk_.u[r] = packbf(kr[kk], kr[kk + 1]);
      }
#pragma unroll
      for (int mt = 0; mt < 4; ++mt)
        acc[mt] = wmma_bf16(aq[mt][ks], bk_, acc[mt]);
    }
#pragma unroll
    for (int mt = 0; mt < 4; ++mt)
#pragma unroll
      for (int r = 0; r < 8; ++r) {
        int mrow = mt * 16 + r + half * 8;
        S[mrow * SROW + nt * 16 + lm] = acc[mt][r] * 0.125f;
      }
  }
  __syncthreads();

  if (tid < 64) {                          // Phase 2: exact causal softmax
    int row = tid, qg = q0 + row;
    float* srow = &S[row * SROW];
    float mx = -1e30f;
    for (int j = 0; j <= qg; ++j) mx = fmaxf(mx, srow[j]);
    float sum = 0.f;
    for (int j = 0; j <= qg; ++j) {
      float e = __expf(srow[j] - mx);
      srow[j] = e;
      sum += e;
    }
    float inv = 1.f / sum;
    for (int j = 0; j <= qg; ++j) srow[j] *= inv;
    for (int j = qg + 1; j < q0 + 64; ++j) srow[j] = 0.f;
  }
  __syncthreads();

  int mt = wid;                            // Phase 3: O = P V
  v8f accO[4];
#pragma unroll
  for (int nt = 0; nt < 4; ++nt) accO[nt] = vzero8();
  int kend = q0 + 64;
  for (int ks2 = 0; ks2 < kend; ks2 += 32) {
    FragBF ap_;
    const float* pr = &S[(mt * 16 + lm) * SROW + ks2];
#pragma unroll
    for (int r = 0; r < 8; ++r) {
      int kk = half * 16 + 2 * r;
      ap_.u[r] = packbf(pr[kk], pr[kk + 1]);
    }
#pragma unroll
    for (int nt = 0; nt < 4; ++nt) {
      FragBF bv_;
#pragma unroll
      for (int r = 0; r < 8; ++r) {
        int kk = ks2 + half * 16 + 2 * r;
        bv_.u[r] = packbf(vp[(size_t)kk * D + nt * 16 + lm],
                          vp[(size_t)(kk + 1) * D + nt * 16 + lm]);
      }
      accO[nt] = wmma_bf16(ap_, bv_, accO[nt]);
    }
  }
#pragma unroll
  for (int nt = 0; nt < 4; ++nt)
#pragma unroll
    for (int r = 0; r < 8; ++r) {
      int gm = q0 + mt * 16 + r + half * 8;
      op[(size_t)gm * D + nt * 16 + lm] = accO[nt][r];
    }
}

// -------------------------------------------------- residual add + LayerNorm
__global__ __launch_bounds__(256) void add_ln_kernel(
    const float* __restrict__ h, const float* __restrict__ res,
    const float* __restrict__ g, const float* __restrict__ bta,
    float* __restrict__ out, unsigned short* __restrict__ ob, int D) {
  __shared__ float s1[256], s2[256];
  int row = blockIdx.x, tid = threadIdx.x;
  const float* hp = h + (size_t)row * D;
  const float* rp = res + (size_t)row * D;
  float sum = 0.f, sq = 0.f;
  for (int i = tid; i < D; i += 256) {
    float vv = hp[i] + rp[i];
    sum += vv;
    sq += vv * vv;
  }
  s1[tid] = sum; s2[tid] = sq;
  __syncthreads();
  for (int off = 128; off > 0; off >>= 1) {
    if (tid < off) { s1[tid] += s1[tid + off]; s2[tid] += s2[tid + off]; }
    __syncthreads();
  }
  float mean = s1[0] / D;
  float var  = s2[0] / D - mean * mean;
  float rstd = rsqrtf(var + 1e-5f);
  for (int i = tid; i < D; i += 256) {
    float vv = hp[i] + rp[i];
    float o = (vv - mean) * rstd * g[i] + bta[i];
    out[(size_t)row * D + i] = o;
    if (ob) ob[(size_t)row * D + i] = f2bf(o);
  }
}

// ------------------------------------------------------------------- host
static void launch_gemm(hipStream_t s, const float* A, const float* Bw,
                        const float* bias, float* C, int M, int N, int K,
                        int lda, int ldb, int ldc, int batch, int Hdiv,
                        long long bsA, long long bsAh, long long bsB,
                        long long bsBh, long long bsC, long long bsCh,
                        int transA, float alpha, int relu) {
  dim3 g((N + BN - 1) / BN, (M + BM - 1) / BM, batch), blk(256);
  gemm_wmma<<<g, blk, 0, s>>>(A, Bw, bias, C, M, N, K, lda, ldb, ldc,
                              bsA, bsAh, bsB, bsBh, bsC, bsCh, Hdiv,
                              transA, 0, alpha, relu);
}

extern "C" void kernel_launch(void* const* d_in, const int* in_sizes, int n_in,
                              void* d_out, int out_size, void* d_ws,
                              size_t ws_size, hipStream_t stream) {
  (void)in_sizes; (void)n_in; (void)out_size; (void)ws_size;

  const int*   enc_seq = (const int*)d_in[0];
  const int*   dec_seq = (const int*)d_in[1];
  const float* tok_e   = (const float*)d_in[2];
  const float* pos_e   = (const float*)d_in[3];
  const float* tok_d   = (const float*)d_in[4];
  const float* pos_d   = (const float*)d_in[5];
  const float* eA[8]; for (int i = 0; i < 8; ++i) eA[i] = (const float*)d_in[6 + i];
  const float* eF[6]; for (int i = 0; i < 6; ++i) eF[i] = (const float*)d_in[14 + i];
  const float* sA[8]; for (int i = 0; i < 8; ++i) sA[i] = (const float*)d_in[20 + i];
  const float* cA[8]; for (int i = 0; i < 8; ++i) cA[i] = (const float*)d_in[28 + i];
  const float* dF[6]; for (int i = 0; i < 6; ++i) dF[i] = (const float*)d_in[36 + i];
  const float* W_out = (const float*)d_in[42];
  const float* b_out = (const float*)d_in[43];
  float* logits = (float*)d_out;

  // ---- workspace: fp32 region then bf16 region
  float* xe  = (float*)d_ws;                       // [2048,1024] enc state
  float* y   = xe  + (size_t)ROWS_E * Dm;          // [2048,1024] dec state
  float* qb  = y   + (size_t)ROWS_D * Dm;
  float* kb  = qb  + (size_t)ROWS_E * Dm;
  float* vb  = kb  + (size_t)ROWS_E * Dm;
  float* t0  = vb  + (size_t)ROWS_E * Dm;          // attn/ffn tmp
  float* mh  = t0  + (size_t)ROWS_E * Dm;          // [B*H,64,64] K^T V
  float* f32end = mh + (size_t)Bb * Hh * HD * HD;

  unsigned short* bfp = (unsigned short*)f32end;
  auto takeb = [&](size_t n) { unsigned short* p = bfp; bfp += n; return p; };
  unsigned short* xe_b  = takeb((size_t)ROWS_E * Dm);
  unsigned short* y_b   = takeb((size_t)ROWS_D * Dm);
  unsigned short* hid_b = takeb((size_t)ROWS_E * DFF);
  // transposed bf16 weights [N][K] per layer slice
  unsigned short* eWq_t = takeb((size_t)Lc * Dm * Dm);
  unsigned short* eWk_t = takeb((size_t)Lc * Dm * Dm);
  unsigned short* eWv_t = takeb((size_t)Lc * Dm * Dm);
  unsigned short* eW1_t = takeb((size_t)Lc * Dm * DFF);
  unsigned short* eW2_t = takeb((size_t)Lc * DFF * Dm);
  unsigned short* sWq_t = takeb((size_t)Lc * Dm * Dm);
  unsigned short* sWk_t = takeb((size_t)Lc * Dm * Dm);
  unsigned short* sWv_t = takeb((size_t)Lc * Dm * Dm);
  unsigned short* cWq_t = takeb((size_t)Lc * Dm * Dm);
  unsigned short* cWk_t = takeb((size_t)Lc * Dm * Dm);
  unsigned short* cWv_t = takeb((size_t)Lc * Dm * Dm);
  unsigned short* dW1_t = takeb((size_t)Lc * Dm * DFF);
  unsigned short* dW2_t = takeb((size_t)Lc * DFF * Dm);
  unsigned short* Wout_t = takeb((size_t)Dm * VD);

  auto convW = [&](const float* w, unsigned short* wt, int K, int N, int L) {
    for (int i = 0; i < L; ++i) {
      dim3 g(N / 32, K / 32), blk(256);
      transpose_convert_kernel<<<g, blk, 0, stream>>>(
          w + (size_t)i * K * N, wt + (size_t)i * K * N, K, N);
    }
  };
  auto linearT = [&](const unsigned short* Ab, const unsigned short* Wt,
                     const float* bias, float* C, unsigned short* Cb,
                     int M, int K, int N, int relu) {
    dim3 g(N / BN, M / BM), blk(256);
    gemm_bf16_tdm<<<g, blk, 0, stream>>>(Ab, Wt, bias, C, Cb, M, N, K, relu);
  };
  auto add_ln = [&](const float* h, const float* r, const float* g,
                    const float* b2, float* o, unsigned short* ob, int rows) {
    add_ln_kernel<<<rows, 256, 0, stream>>>(h, r, g, b2, o, ob, Dm);
  };
  // Linear attention (no softmax): o = Q (K^T V) / sqrt(hd), per (b,h)
  auto lin_attn = [&](const float* qq, const float* kk, const float* vv,
                      float* o, int Tq, int Tkv) {
    launch_gemm(stream, kk, vv, nullptr, mh, HD, HD, Tkv, Dm, Dm, HD,
                Bb * Hh, Hh, (long long)Tkv * Dm, HD,
                (long long)Tkv * Dm, HD,
                (long long)Hh * HD * HD, (long long)HD * HD, 1, 1.0f, 0);
    launch_gemm(stream, qq, mh, nullptr, o, Tq, HD, HD, Dm, HD, Dm,
                Bb * Hh, Hh, (long long)Tq * Dm, HD,
                (long long)Hh * HD * HD, (long long)HD * HD,
                (long long)Tq * Dm, HD, 0, 0.125f, 0);
  };

  (void)hipFuncSetAttribute((const void*)attn_softmax_causal,
                            hipFuncAttributeMaxDynamicSharedMemorySize,
                            (int)ATTN_SMEM);

  // ---- one-time (per launch) weight conversion to transposed bf16
  convW(eA[0], eWq_t, Dm, Dm, Lc); convW(eA[2], eWk_t, Dm, Dm, Lc);
  convW(eA[4], eWv_t, Dm, Dm, Lc);
  convW(eF[0], eW1_t, Dm, DFF, Lc); convW(eF[2], eW2_t, DFF, Dm, Lc);
  convW(sA[0], sWq_t, Dm, Dm, Lc); convW(sA[2], sWk_t, Dm, Dm, Lc);
  convW(sA[4], sWv_t, Dm, Dm, Lc);
  convW(cA[0], cWq_t, Dm, Dm, Lc); convW(cA[2], cWk_t, Dm, Dm, Lc);
  convW(cA[4], cWv_t, Dm, Dm, Lc);
  convW(dF[0], dW1_t, Dm, DFF, Lc); convW(dF[2], dW2_t, DFF, Dm, Lc);
  convW(W_out, Wout_t, Dm, VD, 1);

  // ---------------- encoder ----------------
  embed_kernel<<<ROWS_E, 256, 0, stream>>>(enc_seq, tok_e, pos_e, xe, xe_b,
                                           TE, Dm);
  for (int i = 0; i < Lc; ++i) {
    size_t wo = (size_t)i * Dm * Dm, vo = (size_t)i * Dm;
    linearT(xe_b, eWq_t + wo, eA[1] + vo, qb, nullptr, ROWS_E, Dm, Dm, 0);
    linearT(xe_b, eWk_t + wo, eA[3] + vo, kb, nullptr, ROWS_E, Dm, Dm, 0);
    linearT(xe_b, eWv_t + wo, eA[5] + vo, vb, nullptr, ROWS_E, Dm, Dm, 0);
    lin_attn(qb, kb, vb, t0, TE, TE);
    add_ln(t0, xe, eA[6] + vo, eA[7] + vo, xe, xe_b, ROWS_E);
    linearT(xe_b, eW1_t + (size_t)i * Dm * DFF, eF[1] + (size_t)i * DFF,
            nullptr, hid_b, ROWS_E, Dm, DFF, 1);
    linearT(hid_b, eW2_t + (size_t)i * DFF * Dm, eF[3] + vo, t0, nullptr,
            ROWS_E, DFF, Dm, 0);
    add_ln(t0, xe, eF[4] + vo, eF[5] + vo, xe, xe_b, ROWS_E);
  }

  // ---------------- decoder ----------------
  embed_kernel<<<ROWS_D, 256, 0, stream>>>(dec_seq, tok_d, pos_d, y, y_b,
                                           TD, Dm);
  for (int i = 0; i < Lc; ++i) {
    size_t wo = (size_t)i * Dm * Dm, vo = (size_t)i * Dm;
    // causal self-attention (softmax)
    linearT(y_b, sWq_t + wo, sA[1] + vo, qb, nullptr, ROWS_D, Dm, Dm, 0);
    linearT(y_b, sWk_t + wo, sA[3] + vo, kb, nullptr, ROWS_D, Dm, Dm, 0);
    linearT(y_b, sWv_t + wo, sA[5] + vo, vb, nullptr, ROWS_D, Dm, Dm, 0);
    attn_softmax_causal<<<dim3(TD / 64, Hh, Bb), 128, ATTN_SMEM, stream>>>(
        qb, kb, vb, t0, TD, Dm);
    add_ln(t0, y, sA[6] + vo, sA[7] + vo, y, y_b, ROWS_D);
    // cross-attention (linear: no softmax in reference)
    linearT(y_b, cWq_t + wo, cA[1] + vo, qb, nullptr, ROWS_D, Dm, Dm, 0);
    linearT(xe_b, cWk_t + wo, cA[3] + vo, kb, nullptr, ROWS_E, Dm, Dm, 0);
    linearT(xe_b, cWv_t + wo, cA[5] + vo, vb, nullptr, ROWS_E, Dm, Dm, 0);
    lin_attn(qb, kb, vb, t0, TD, TE);
    add_ln(t0, y, cA[6] + vo, cA[7] + vo, y, y_b, ROWS_D);
    // ffn
    linearT(y_b, dW1_t + (size_t)i * Dm * DFF, dF[1] + (size_t)i * DFF,
            nullptr, hid_b, ROWS_D, Dm, DFF, 1);
    linearT(hid_b, dW2_t + (size_t)i * DFF * Dm, dF[3] + vo, t0, nullptr,
            ROWS_D, DFF, Dm, 0);
    add_ln(t0, y, dF[4] + vo, dF[5] + vo, y, y_b, ROWS_D);
  }

  // ---------------- output projection ----------------
  linearT(y_b, Wout_t, b_out, logits, nullptr, ROWS_D, Dm, VD, 0);
}
